// BioNet_62491774157015
// MI455X (gfx1250) — compile-verified, matrix operand order checked by reading, not run on hardware
//
#include <hip/hip_runtime.h>

// ---------------------------------------------------------------------------
// BioNet steady-state dynamics on MI455X (gfx1250)
//   X_{t+1} = mml(W @ X_t + X_bias),  120 steps, W: 4096x4096, X: 4096x512
// W -> f16 once (64MB f32 -> 32MB f16, L2-resident), state batch-major f16,
// 119 sequential WMMA-GEMM step kernels (step 1 is just mml(X_bias)).
// All global accesses via explicit address_space(1) so the backend emits
// global_load_b128/global_store_b128 (not flat_*), single base address per
// matrix with immediate offsets for the fragment fan-out.
// ---------------------------------------------------------------------------

typedef __attribute__((ext_vector_type(16))) _Float16 v16h;
typedef __attribute__((ext_vector_type(8)))  float    v8f;
typedef __attribute__((ext_vector_type(4)))  float    f32x4;

typedef const _Float16 __attribute__((address_space(1))) gc_half;
typedef const float    __attribute__((address_space(1))) gc_float;
typedef const f32x4    __attribute__((address_space(1))) gc_f32x4;
typedef f32x4          __attribute__((address_space(1))) g_f32x4;
typedef _Float16       __attribute__((address_space(1))) g_half;
typedef float          __attribute__((address_space(1))) g_float;

#define NN    4096   // nodes (M and K of the GEMM)
#define NB    512    // batch (N of the GEMM)
#define STEPS 120
#define BM    128    // block tile M
#define BN    128    // block tile N
// 8 waves per block: 2 (M) x 4 (N), each wave computes 64x32 (4x2 WMMA tiles)

// Branchless mml: v_cndmask selects + v_rcp_f32 (no precise-div expansion).
__device__ __forceinline__ float mml_f(float x) {
    float y = (x < 0.0f) ? 0.01f * x : x;
    float r = 1.0f - 0.25f * __builtin_amdgcn_rcpf(__builtin_fmaxf(x, 0.5f));
    return (x > 0.5f) ? r : y;
}

// 16-bit A(16x32)/B(32x16) fragment: lane (lo,hi) holds K = hi*8 + [0..7]
// and K = 16 + hi*8 + [0..7]; two 16B global loads, 32B apart.
__device__ __forceinline__ v16h load_frag16(gc_half* p) {
    union { v16h h; f32x4 f[2]; } u;
    gc_f32x4* q = (gc_f32x4*)p;
    u.f[0] = q[0];   // halves [0..7]
    u.f[1] = q[2];   // halves [16..23]
    return u.h;
}

// --- one-time: f32 weights -> f16 (8 elements/thread, b128 traffic) --------
__global__ __launch_bounds__(256) void cvt_weights(const float* __restrict__ W,
                                                   _Float16* __restrict__ Wh) {
    size_t i = ((size_t)blockIdx.x * blockDim.x + threadIdx.x) * 8;
    gc_f32x4* src = (gc_f32x4*)(W + i);
    f32x4 a = src[0], b = src[1];
    union { f32x4 f; _Float16 h[8]; } u;
    u.h[0] = (_Float16)a[0]; u.h[1] = (_Float16)a[1];
    u.h[2] = (_Float16)a[2]; u.h[3] = (_Float16)a[3];
    u.h[4] = (_Float16)b[0]; u.h[5] = (_Float16)b[1];
    u.h[6] = (_Float16)b[2]; u.h[7] = (_Float16)b[3];
    *(g_f32x4*)(Wh + i) = u.f;
}

// --- one-time: X_bias (batch-major) and step-1 state  X1 = mml(X_bias) -----
__global__ __launch_bounds__(256) void init_state(const float* __restrict__ Xfull,
                                                  const float* __restrict__ bias,
                                                  float* __restrict__ Xbias,
                                                  _Float16* __restrict__ X1) {
    size_t i = ((size_t)blockIdx.x * blockDim.x + threadIdx.x) * 4;  // over NB*NN
    int m = (int)(i & (NN - 1));
    f32x4 x = *(gc_f32x4*)(Xfull + i);
    f32x4 xb;
    union { unsigned long long u64; _Float16 h[4]; } hh;
#pragma unroll
    for (int r = 0; r < 4; ++r) {
        xb[r] = x[r] + ((gc_float*)bias)[m + r];
        hh.h[r] = (_Float16)mml_f(xb[r]);
    }
    *(g_f32x4*)(Xbias + i) = xb;
    *(unsigned long long __attribute__((address_space(1)))*)(X1 + i) = hh.u64;
}

// --- per-step: Xnew = mml(W @ Xold + Xbias); optionally emit f32 output ----
__global__ __launch_bounds__(256)
void bionet_step(const _Float16* __restrict__ Wh,     // [NN][NN] row-major f16
                 const _Float16* __restrict__ Xold,   // [NB][NN] batch-major f16
                 const float*    __restrict__ Xbias,  // [NB][NN] batch-major f32
                 _Float16*       __restrict__ Xnew,   // [NB][NN] batch-major f16
                 float*          __restrict__ outp)   // [NB][NN] f32 (last step)
{
    const int lane = threadIdx.x & 31;
    const int wave = threadIdx.x >> 5;
    const int lo   = lane & 15;
    const int hi   = lane >> 4;
    const int wm   = wave >> 2;                 // 0..1
    const int wn   = wave & 3;                  // 0..3
    const int m0   = blockIdx.x * BM + wm * 64; // wave M origin
    const int n0   = blockIdx.y * BN + wn * 32; // wave N origin

    const v8f vzero = {0.f, 0.f, 0.f, 0.f, 0.f, 0.f, 0.f, 0.f};
    v8f acc[4][2];
#pragma unroll
    for (int i = 0; i < 4; ++i)
#pragma unroll
        for (int j = 0; j < 2; ++j) acc[i][j] = vzero;

    // One base address per matrix; fragment fan-out via immediate offsets
    // (i*16*NN halves = 128KB, within the signed 24-bit IOFFSET range).
    gc_half* pA = (gc_half*)Wh   + (size_t)(m0 + lo) * NN + hi * 8;
    gc_half* pB = (gc_half*)Xold + (size_t)(n0 + lo) * NN + hi * 8;

    for (int k = 0; k < NN; k += 32) {
        v16h a[4], b[2];
#pragma unroll
        for (int j = 0; j < 2; ++j) b[j] = load_frag16(pB + k + j * 16 * NN);
#pragma unroll
        for (int i = 0; i < 4; ++i) a[i] = load_frag16(pA + k + i * 16 * NN);
#pragma unroll
        for (int i = 0; i < 4; ++i)
#pragma unroll
            for (int j = 0; j < 2; ++j)
                acc[i][j] = __builtin_amdgcn_wmma_f32_16x16x32_f16(
                    /*neg_a=*/false, a[i], /*neg_b=*/false, b[j],
                    /*c_mod=*/(short)0, acc[i][j],
                    /*reuse_a=*/false, /*reuse_b=*/false);
    }

    // Epilogue: C/D layout -> lane holds (M = r + hi*8, N = lo), r = 0..7,
    // i.e. 8 consecutive m per lane in batch-major memory.
#pragma unroll
    for (int i = 0; i < 4; ++i) {
        const int mbase = m0 + i * 16 + hi * 8;
#pragma unroll
        for (int j = 0; j < 2; ++j) {
            const int n = n0 + j * 16 + lo;
            const size_t off = (size_t)n * NN + mbase;
            gc_f32x4* xb = (gc_f32x4*)(Xbias + off);
            f32x4 x0 = xb[0], x1 = xb[1];
            union { f32x4 f; _Float16 h[8]; } hs;
            f32x4 o0, o1;
#pragma unroll
            for (int r = 0; r < 8; ++r) {
                float xbias_r = (r < 4) ? x0[r & 3] : x1[r & 3];
                float v = mml_f(acc[i][j][r] + xbias_r);
                if (r < 4) o0[r & 3] = v; else o1[r & 3] = v;
                hs.h[r] = (_Float16)v;
            }
            *(g_f32x4*)(Xnew + off) = hs.f;      // 16B f16 state store
            if (outp) {
                g_f32x4* op = (g_f32x4*)(outp + off);
                op[0] = o0;
                op[1] = o1;
            }
        }
    }
}

extern "C" void kernel_launch(void* const* d_in, const int* in_sizes, int n_in,
                              void* d_out, int out_size, void* d_ws, size_t ws_size,
                              hipStream_t stream) {
    const float* Xfull = (const float*)d_in[0];  // (512, 4096) f32
    const float* W     = (const float*)d_in[1];  // (4096, 4096) f32
    const float* bias  = (const float*)d_in[2];  // (4096, 1) f32
    float* out = (float*)d_out;                  // (512, 4096) f32

    char* ws = (char*)d_ws;
    const size_t W_H16_BYTES  = (size_t)NN * NN * sizeof(_Float16);  // 32 MB
    const size_t XBIAS_BYTES  = (size_t)NB * NN * sizeof(float);     //  8 MB
    const size_t STATE_BYTES  = (size_t)NB * NN * sizeof(_Float16);  //  4 MB
    _Float16* Wh    = (_Float16*)ws;
    float*    Xbias = (float*)(ws + W_H16_BYTES);
    _Float16* Xa    = (_Float16*)(ws + W_H16_BYTES + XBIAS_BYTES);
    _Float16* Xb    = (_Float16*)(ws + W_H16_BYTES + XBIAS_BYTES + STATE_BYTES);

    // One-time prep (re-done every call for determinism; cheap vs 119 GEMMs).
    cvt_weights<<<((size_t)NN * NN) / (256 * 8), 256, 0, stream>>>(W, Wh);
    init_state<<<((size_t)NB * NN) / (256 * 4), 256, 0, stream>>>(Xfull, bias, Xbias, Xa);

    // Steps 2..120 (step 1 == init_state since X0 = 0).
    _Float16* cur = Xa;
    _Float16* nxt = Xb;
    dim3 grid(NN / BM, NB / BN);  // 32 x 4 blocks
    for (int s = 1; s < STEPS; ++s) {
        float* op = (s == STEPS - 1) ? out : nullptr;
        bionet_step<<<grid, 256, 0, stream>>>(Wh, cur, Xbias, nxt, op);
        _Float16* t = cur; cur = nxt; nxt = t;
    }
}
